// EnhancedGraphAttentionLayer_36653250904537
// MI455X (gfx1250) — matrix operand order repaired and checked by reading, not compiled, not run on hardware
//
#include <hip/hip_runtime.h>

#define ALPHA    0.2f
#define LN_EPS   1e-5f
#define MASK_VAL -1000000000.0f
#define NN 1024
#define DD 64

typedef __attribute__((ext_vector_type(16))) _Float16 v16h;
typedef __attribute__((ext_vector_type(8)))  float    v8f;

__device__ __forceinline__ float lrelu(float x) { return fmaxf(x, ALPHA * x); }

// ---------------------------------------------------------------------------
// Phase 1a: Wh = h@W ; ei = h@E_i ; ej = h@E_j      (all 1024x64 @ 64x64, fp32)
// ---------------------------------------------------------------------------
__global__ __launch_bounds__(256) void gat_pre1(
    const float* __restrict__ h, const float* __restrict__ W,
    const float* __restrict__ edge_w,
    float* __restrict__ Wh, float* __restrict__ ei, float* __restrict__ ej)
{
    const int idx = blockIdx.x * blockDim.x + threadIdx.x;   // 1024*64 threads
    const int row = idx >> 6, col = idx & 63;
    const float* hr = h + row * DD;
    float a = 0.f, b = 0.f, c = 0.f;
#pragma unroll 8
    for (int k = 0; k < DD; ++k) {
        const float hv = hr[k];
        a = fmaf(hv, W[k * DD + col], a);
        b = fmaf(hv, edge_w[k * DD + col], b);            // E_i = edge_w[:64]
        c = fmaf(hv, edge_w[(DD + k) * DD + col], c);     // E_j = edge_w[64:]
    }
    Wh[idx] = a; ei[idx] = b; ej[idx] = c;
}

// ---------------------------------------------------------------------------
// Phase 1b: ai = Wh@A_i ; aj = Wh@A_j
// ---------------------------------------------------------------------------
__global__ __launch_bounds__(256) void gat_pre2(
    const float* __restrict__ Wh, const float* __restrict__ attn_w1,
    float* __restrict__ ai, float* __restrict__ aj)
{
    const int idx = blockIdx.x * blockDim.x + threadIdx.x;
    const int row = idx >> 6, col = idx & 63;
    const float* wr = Wh + row * DD;
    float a = 0.f, b = 0.f;
#pragma unroll 8
    for (int k = 0; k < DD; ++k) {
        const float wv = wr[k];
        a = fmaf(wv, attn_w1[k * DD + col], a);           // A_i = attn_w1[:64]
        b = fmaf(wv, attn_w1[(DD + k) * DD + col], b);    // A_j = attn_w1[64:128]
    }
    ai[idx] = a; aj[idx] = b;
}

// ---------------------------------------------------------------------------
// Phase 2 (dominant, 8.6 GFLOP): fused edge-score kernel.
// Wave w handles row i; per 16-wide j tile:
//   T[m,k]  = lrelu(ei[i,k] + ej[j0+m,k] + eb[k])           (f32 -> f16)
//   U       = T @ A_e  + (ai[i]+aj[j]+b1)   via v_wmma_f32_16x16x32_f16,
//             bias folded into the C accumulator init.
//   e[i,j]  = sum_n lrelu(U[m,n]) * w2[n] + b2              (shfl_xor reduce)
// ---------------------------------------------------------------------------
__global__ __launch_bounds__(256) void gat_edge_scores(
    const float* __restrict__ ei, const float* __restrict__ ej,
    const float* __restrict__ ai, const float* __restrict__ aj,
    const float* __restrict__ Ae, const float* __restrict__ eb,
    const float* __restrict__ b1, const float* __restrict__ w2,
    const float* __restrict__ b2p, float* __restrict__ e)
{
    const int lane = threadIdx.x & 31;
    const int wave = threadIdx.x >> 5;
    const int i    = blockIdx.x * 8 + wave;          // 128 blocks * 8 waves = 1024 rows
    const int half = lane >> 4;                      // 0 | 1
    const int col  = lane & 15;                      // N-index / A-row index
    const int kb   = half * 8;                       // A-layout K base for this half
    const float b2 = b2p[0];

    // B = A_e resident in registers, ISA B-layout: element c -> K = half*8 + c + (c>=8 ? 8 : 0),
    // column N = col. [n-tile][k-chunk]
    v16h B[4][2];
#pragma unroll
    for (int nt = 0; nt < 4; ++nt)
#pragma unroll
        for (int c0 = 0; c0 < 2; ++c0)
#pragma unroll
            for (int c = 0; c < 16; ++c) {
                const int k = c0 * 32 + half * 8 + c + ((c >= 8) ? 8 : 0);
                B[nt][c0][c] = (_Float16)Ae[k * DD + nt * 16 + col];
            }

    // s[k] = ei[i,k] + edge_b[k] for exactly the K slots this lane packs into A.
    float sA[32];
#pragma unroll
    for (int c0 = 0; c0 < 2; ++c0)
#pragma unroll
        for (int c = 0; c < 16; ++c) {
            const int k = c0 * 32 + kb + c + ((c >= 8) ? 8 : 0);
            sA[c0 * 16 + c] = ei[i * DD + k] + eb[k];
        }

    float aib[4], w2l[4];
#pragma unroll
    for (int nt = 0; nt < 4; ++nt) {
        aib[nt] = ai[i * DD + nt * 16 + col] + b1[nt * 16 + col];
        w2l[nt] = w2[nt * 16 + col];
    }

    for (int j0 = 0; j0 < NN; j0 += 16) {
        // Build A (two K=32 chunks) in ISA A-layout; A row M = col -> j = j0+col.
        const float* ejrow = ej + (j0 + col) * DD;
        v16h A0, A1;
#pragma unroll
        for (int c = 0; c < 16; ++c) {
            const int k0 = kb + c + ((c >= 8) ? 8 : 0);
            A0[c] = (_Float16)lrelu(sA[c]      + ejrow[k0]);
            A1[c] = (_Float16)lrelu(sA[16 + c] + ejrow[32 + k0]);
        }

        // C init = aj[j0+m] + ai[i] + b1  in C/D layout (VGPR r -> M = r+8*half).
        v8f C[4];
#pragma unroll
        for (int nt = 0; nt < 4; ++nt)
#pragma unroll
            for (int r = 0; r < 8; ++r)
                C[nt][r] = aj[(j0 + r + 8 * half) * DD + nt * 16 + col] + aib[nt];

#pragma unroll
        for (int nt = 0; nt < 4; ++nt) {
            C[nt] = __builtin_amdgcn_wmma_f32_16x16x32_f16(
                false, A0, false, B[nt][0], (short)0, C[nt], false, false);
            C[nt] = __builtin_amdgcn_wmma_f32_16x16x32_f16(
                false, A1, false, B[nt][1], (short)0, C[nt], false, false);
        }

        // lrelu, weight by w2, reduce the 64 N-columns across the 16-lane half.
        float part[8];
#pragma unroll
        for (int r = 0; r < 8; ++r) {
            float acc = 0.f;
#pragma unroll
            for (int nt = 0; nt < 4; ++nt)
                acc = fmaf(lrelu(C[nt][r]), w2l[nt], acc);
            acc += __shfl_xor(acc, 1);
            acc += __shfl_xor(acc, 2);
            acc += __shfl_xor(acc, 4);
            acc += __shfl_xor(acc, 8);
            part[r] = acc;
        }
        if (col == 0) {
#pragma unroll
            for (int r = 0; r < 8; ++r)
                e[i * NN + j0 + half * 8 + r] = part[r] + b2;
        }
    }
}

// ---------------------------------------------------------------------------
// Phase 3: mask + softmax over j, h' = attn@Wh + h, LayerNorm. One block/row.
// ---------------------------------------------------------------------------
__global__ __launch_bounds__(256) void gat_softmax_ln(
    const float* __restrict__ e, const int* __restrict__ adj,
    const float* __restrict__ Wh, const float* __restrict__ h,
    const float* __restrict__ ln_g, const float* __restrict__ ln_b,
    float* __restrict__ out)
{
    __shared__ float p[NN];
    __shared__ float red[256];
    __shared__ float stats[2];
    const int i = blockIdx.x;
    const int t = threadIdx.x;

    float vals[4];
    float lmax = -3.4e38f;
#pragma unroll
    for (int q = 0; q < 4; ++q) {
        const int j = t + q * 256;
        float v = e[i * NN + j];
        v = (adj[i * NN + j] != 0) ? v : MASK_VAL;
        vals[q] = v;
        lmax = fmaxf(lmax, v);
    }
    red[t] = lmax; __syncthreads();
    for (int s = 128; s > 0; s >>= 1) { if (t < s) red[t] = fmaxf(red[t], red[t + s]); __syncthreads(); }
    const float rmax = red[0]; __syncthreads();

    float lsum = 0.f;
#pragma unroll
    for (int q = 0; q < 4; ++q) {
        const float pv = __expf(vals[q] - rmax);
        p[t + q * 256] = pv;
        lsum += pv;
    }
    red[t] = lsum; __syncthreads();
    for (int s = 128; s > 0; s >>= 1) { if (t < s) red[t] += red[t + s]; __syncthreads(); }
    const float rsum = red[0]; __syncthreads();

    // aggregation: thread (g,d) sums quarter g of j for feature d
    const int d = t & 63, g = t >> 6;
    float acc = 0.f;
    const float* Whc = Wh + d;
    for (int j = g * 256; j < g * 256 + 256; ++j)
        acc = fmaf(p[j], Whc[j * DD], acc);
    red[t] = acc; __syncthreads();

    if (t < 64)
        p[t] = (red[t] + red[t + 64] + red[t + 128] + red[t + 192]) / rsum + h[i * DD + t];
    __syncthreads();

    if (t == 0) {
        float mu = 0.f;
        for (int k = 0; k < 64; ++k) mu += p[k];
        mu *= (1.f / 64.f);
        float var = 0.f;
        for (int k = 0; k < 64; ++k) { const float dv = p[k] - mu; var = fmaf(dv, dv, var); }
        var *= (1.f / 64.f);
        stats[0] = mu;
        stats[1] = rsqrtf(var + LN_EPS);
    }
    __syncthreads();

    if (t < 64)
        out[i * DD + t] = (p[t] - stats[0]) * stats[1] * ln_g[t] + ln_b[t];
}

// ---------------------------------------------------------------------------
extern "C" void kernel_launch(void* const* d_in, const int* in_sizes, int n_in,
                              void* d_out, int out_size, void* d_ws, size_t ws_size,
                              hipStream_t stream)
{
    const float* h       = (const float*)d_in[0];
    const int*   adj     = (const int*)  d_in[1];
    const float* W       = (const float*)d_in[2];
    const float* attn_w1 = (const float*)d_in[3];
    const float* attn_b1 = (const float*)d_in[4];
    const float* attn_w2 = (const float*)d_in[5];
    const float* attn_b2 = (const float*)d_in[6];
    const float* edge_w  = (const float*)d_in[7];
    const float* edge_b  = (const float*)d_in[8];
    const float* ln_g    = (const float*)d_in[9];
    const float* ln_b    = (const float*)d_in[10];
    float* out = (float*)d_out;

    float* ws = (float*)d_ws;
    float* Wh = ws;                 // 1024*64
    float* ei = Wh + NN * DD;       // 1024*64
    float* ej = ei + NN * DD;       // 1024*64
    float* ai = ej + NN * DD;       // 1024*64
    float* aj = ai + NN * DD;       // 1024*64
    float* e  = aj + NN * DD;       // 1024*1024

    gat_pre1<<<(NN * DD) / 256, 256, 0, stream>>>(h, W, edge_w, Wh, ei, ej);
    gat_pre2<<<(NN * DD) / 256, 256, 0, stream>>>(Wh, attn_w1, ai, aj);
    gat_edge_scores<<<NN / 8, 256, 0, stream>>>(
        ei, ej, ai, aj, attn_w1 + 2 * DD * DD, edge_b, attn_b1, attn_w2, attn_b2, e);
    gat_softmax_ln<<<NN, 256, 0, stream>>>(e, adj, Wh, h, ln_g, ln_b, out);
}
